// EquivariantProductBlock_6605659702018
// MI455X (gfx1250) — matrix operand order repaired and checked by reading, not compiled, not run on hardware
//
#include <hip/hip_runtime.h>
#include <hip/hip_bf16.h>

// ---------------------------------------------------------------------------
// MACE EquivariantProductBlock on gfx1250 (MI455X), wave32 + WMMA bf16.
//
// Math restructuring: the nested MACE contraction collapses to, per (b,c,m):
//   final = sum_{p,i} A3[c,m,p,i] y[p] x[i] + sum_p A2[c,m,p] y[p] + sum_i A1[c,m,i] x[i]
// with y[(d0,d1)] = x[d0]*x[d1].  Using augmented features f=[y(81),1] (pad 96)
// and output columns j=m*10+i (i<9: cubic+A1 via const feature; i==9: quadratic),
// the whole thing is Z[N x 40] = F[N x 96] @ G[c][96 x 48] (pad) — one small-K
// GEMM per channel -> v_wmma_f32_16x16x32_bf16, then a tiny epilogue.
// G-fragments are staged into LDS with GLOBAL_LOAD_ASYNC_TO_LDS_B128 (ASYNCcnt).
// ---------------------------------------------------------------------------

typedef __attribute__((ext_vector_type(8)))  float  v8f;
typedef __attribute__((ext_vector_type(16))) __bf16 v16bf;

#define C_CH 128
#define NQ   96      // padded K (81 y-products + 1 const + 14 zero)
#define NJ   48      // padded output cols (40 used)
#define INV_SQRT_C 0.08838834764831845f

struct U4x2 { uint4 lo, hi; };

static __device__ __forceinline__ unsigned short f2bf(float f) {
  unsigned u = __builtin_bit_cast(unsigned, f);
  u += 0x7fffu + ((u >> 16) & 1u);        // round-to-nearest-even
  return (unsigned short)(u >> 16);
}

static __device__ __forceinline__ v16bf make_frag(uint4 lo, uint4 hi) {
  U4x2 t; t.lo = lo; t.hi = hi;
  return __builtin_bit_cast(v16bf, t);
}

static __device__ __forceinline__ unsigned pk2(float a, float b) {
  return (unsigned)f2bf(a) | ((unsigned)f2bf(b) << 16);
}

// ---------------------------------------------------------------------------
// Kernel 1: fold U3/U2/U1 with W3/W2/W1 into per-channel G[96][48], store in
// WMMA B-fragment order: gfrag[c][ks][nt][lane][e], bf16.
// B-frag layout (32x16 bf16): lane holds col n=lane%16, element e <-> K =
// e + (lane>=16)*16 within the 32-K step.
// ---------------------------------------------------------------------------
__global__ __launch_bounds__(256)
void mace_pack_g_kernel(const float* __restrict__ u3_s, const float* __restrict__ u2_s,
                        const float* __restrict__ u1_s, const float* __restrict__ u3_v,
                        const float* __restrict__ u2_v, const float* __restrict__ u1_v,
                        const float* __restrict__ w3_s, const float* __restrict__ w2_s,
                        const float* __restrict__ w1_s, const float* __restrict__ w3_v,
                        const float* __restrict__ w2_v, const float* __restrict__ w1_v,
                        unsigned short* __restrict__ gfrag) {
  __shared__ float G[NQ * NJ];
  const int c = blockIdx.x;
  for (int t = threadIdx.x; t < NQ * NJ; t += 256) {
    int q = t / NJ, j = t - q * NJ;
    float val = 0.0f;
    if (j < 40) {
      int m = j / 10, i = j - m * 10;
      if (i < 9) {
        if (q < 81) {                       // cubic: G[q,(m,i)] = sum_k U3*W3
          int d0 = q / 9, d1 = q - d0 * 9;
          const float* U3; const float* W3;
          if (m == 0) { U3 = u3_s + (size_t)((d0 * 9 + d1) * 9 + i) * 4; W3 = w3_s; }
          else        { U3 = u3_v + (size_t)((((m - 1) * 9 + d0) * 9 + d1) * 9 + i) * 4; W3 = w3_v; }
#pragma unroll
          for (int k = 0; k < 4; ++k) val += U3[k] * W3[k * C_CH + c];
        } else if (q == 81) {               // linear via const feature
          val = (m == 0) ? u1_s[i] * w1_s[c]
                         : u1_v[(m - 1) * 9 + i] * w1_v[c];
        }
      } else {                              // i==9: quadratic slot
        if (q < 81) {
          int d0 = q / 9, d1 = q - d0 * 9;
          const float* U2; const float* W2;
          if (m == 0) { U2 = u2_s + (size_t)(d0 * 9 + d1) * 3; W2 = w2_s; }
          else        { U2 = u2_v + (size_t)(((m - 1) * 9 + d0) * 9 + d1) * 3; W2 = w2_v; }
#pragma unroll
          for (int k = 0; k < 3; ++k) val += U2[k] * W2[k * C_CH + c];
        }
      }
    }
    G[t] = val;
  }
  __syncthreads();
  // re-emit in B-fragment order
  for (int t = threadIdx.x; t < 4608; t += 256) {   // 3 ks * 3 nt * 32 lanes * 16 e
    int e = t & 15, lane = (t >> 4) & 31;
    int nt = (t / 512) % 3, ks = t / 1536;
    int lk = e + (lane >> 4) * 16;
    int q  = ks * 32 + lk;
    int n  = nt * 16 + (lane & 15);
    gfrag[(size_t)c * 4608 + t] = f2bf(G[q * NJ + n]);
  }
}

// ---------------------------------------------------------------------------
// Kernel 2: pack wl0 / wl1 (scaled by 1/sqrt(C)) into B-fragment order, K=128.
// ---------------------------------------------------------------------------
__global__ __launch_bounds__(256)
void mace_pack_wl_kernel(const float* __restrict__ wl0, const float* __restrict__ wl1,
                         unsigned short* __restrict__ wf0, unsigned short* __restrict__ wf1) {
  int t = blockIdx.x * 256 + threadIdx.x;     // 32768 total
  int which = t >> 14;
  int r = t & 16383;
  int e = r & 15, lane = (r >> 4) & 31, nt = (r >> 9) & 7, ks = r >> 12;
  int k = ks * 32 + e + (lane >> 4) * 16;
  int n = nt * 16 + (lane & 15);
  const float* W = which ? wl1 : wl0;
  unsigned short v = f2bf(W[k * C_CH + n] * INV_SQRT_C);
  if (which) wf1[r] = v; else wf0[r] = v;
}

// ---------------------------------------------------------------------------
// Kernel 3 (main): one workgroup = (channel c, 128-node block). 8 waves, each
// owns a 16-node WMMA row-tile. 9 WMMAs/wave (3 K-steps x 3 N-tiles).
// ---------------------------------------------------------------------------
__global__ __launch_bounds__(256)
void mace_contract_kernel(const float* __restrict__ x_in,
                          const unsigned short* __restrict__ gfrag,
                          float* __restrict__ sv) {
  __shared__ float          lds_x[128 * 12];        // x, row-padded to 12
  __shared__ unsigned short lds_y[128 * NQ];        // bf16 augmented features
  __shared__ unsigned short lds_g[4608];            // G[c] fragments
  __shared__ float          lds_z[128 * 48];        // WMMA output Z

  const int tid = threadIdx.x;
  const int c   = blockIdx.y;
  const int nb  = blockIdx.x * 128;

  // stage G[c] fragments (9216 B) straight into LDS, no VGPR round-trip:
  // GLOBAL_LOAD_ASYNC_TO_LDS_B128 (GVS mode), tracked by ASYNCcnt.
  {
    unsigned long long gbase = (unsigned long long)(const void*)(gfrag + (size_t)c * 4608);
    unsigned lbase = (unsigned)(uintptr_t)(void*)(&lds_g[0]);
    for (int t = tid; t < 576; t += 256) {          // 576 x 16 B
      unsigned off = (unsigned)t * 16u;
      asm volatile("global_load_async_to_lds_b128 %0, %1, %2"
                   :: "v"(lbase + off), "v"(off), "s"(gbase)
                   : "memory");
    }
  }

  for (int t = tid; t < 128 * 9; t += 256) {
    int node = t / 9, i = t - node * 9;
    lds_x[node * 12 + i] = x_in[(size_t)(nb + node) * (C_CH * 9) + c * 9 + i];
  }
  // async-copy must be complete before any wave signals the barrier
  asm volatile("s_wait_asynccnt 0x0" ::: "memory");
  __syncthreads();

  for (int t = tid; t < 128 * NQ; t += 256) {       // y[p] = x[d0]*x[d1]
    int node = t / NQ, q = t - node * NQ;
    float v;
    if (q < 81) { int d0 = q / 9, d1 = q - d0 * 9;
                  v = lds_x[node * 12 + d0] * lds_x[node * 12 + d1]; }
    else v = (q == 81) ? 1.0f : 0.0f;
    lds_y[t] = f2bf(v);
  }
  __syncthreads();

  const int lane = tid & 31, wave = tid >> 5;
  const int l16 = lane & 15, lh = lane >> 4;
  const int node = wave * 16 + l16;

  // A-frags: lane holds row M=lane%16; elements 0-15 -> K = q0..q0+15,
  // q0 = ks*32 + (lane>=16)*16.
  v16bf afrag[3];
#pragma unroll
  for (int ks = 0; ks < 3; ++ks) {
    int q0 = ks * 32 + lh * 16;
    uint4 lo = *(const uint4*)&lds_y[node * NQ + q0];
    uint4 hi = *(const uint4*)&lds_y[node * NQ + q0 + 8];
    afrag[ks] = make_frag(lo, hi);
  }

  v8f acc[3] = {};
#pragma unroll
  for (int nt = 0; nt < 3; ++nt) {
#pragma unroll
    for (int ks = 0; ks < 3; ++ks) {
      const uint4* p = (const uint4*)&lds_g[((ks * 3 + nt) * 32 + lane) * 16];
      v16bf bfrag = make_frag(p[0], p[1]);
      acc[nt] = __builtin_amdgcn_wmma_f32_16x16x32_bf16(
          false, afrag[ks], false, bfrag, (short)0, acc[nt], false, false);
    }
  }

  // scatter Z (D-layout: VGPR r -> row r / r+8, col = lane%16)
#pragma unroll
  for (int nt = 0; nt < 3; ++nt)
#pragma unroll
    for (int r = 0; r < 8; ++r)
      lds_z[(wave * 16 + r + lh * 8) * 48 + nt * 16 + l16] = acc[nt][r];
  __syncthreads();

  // epilogue: final = Z[m,9] + sum_i Z[m,i]*x[i]  -> sv[b][m][c]
  for (int t = tid; t < 512; t += 256) {
    int nd = t >> 2, m = t & 3;
    const float* z  = &lds_z[nd * 48 + m * 10];
    const float* xx = &lds_x[nd * 12];
    float f = z[9];
#pragma unroll
    for (int i = 0; i < 9; ++i) f += z[i] * xx[i];
    sv[((size_t)(nb + nd) * 4 + m) * C_CH + c] = f;
  }
}

// ---------------------------------------------------------------------------
// Kernel 4: equivariant linear via WMMA (K=128 channels, 4 k-steps),
// y0 = s@wl0, y1[d,m] = sum_c v[c,m] wl1[c,d]; add sc; interleaved output.
// ---------------------------------------------------------------------------
__global__ __launch_bounds__(256)
void mace_linear_kernel(const float* __restrict__ sv,
                        const unsigned short* __restrict__ wf0,
                        const unsigned short* __restrict__ wf1,
                        const float* __restrict__ sc,
                        float* __restrict__ out) {
  const int tid = threadIdx.x;
  const int lane = tid & 31, wave = tid >> 5;
  const int l16 = lane & 15, lh = lane >> 4;
  const int tile = (blockIdx.x * 8 + wave) * 16;
  const int node = tile + l16;

#pragma unroll 1
  for (int m = 0; m < 4; ++m) {
    v16bf afr[4];
#pragma unroll
    for (int ks = 0; ks < 4; ++ks) {
      int c0 = ks * 32 + lh * 16;
      const float* base = sv + ((size_t)node * 4 + m) * C_CH;
      float4 a0 = *(const float4*)(base + c0);
      float4 a1 = *(const float4*)(base + c0 + 4);
      float4 b0 = *(const float4*)(base + c0 + 8);
      float4 b1 = *(const float4*)(base + c0 + 12);
      uint4 lo = make_uint4(pk2(a0.x, a0.y), pk2(a0.z, a0.w),
                            pk2(a1.x, a1.y), pk2(a1.z, a1.w));
      uint4 hi = make_uint4(pk2(b0.x, b0.y), pk2(b0.z, b0.w),
                            pk2(b1.x, b1.y), pk2(b1.z, b1.w));
      afr[ks] = make_frag(lo, hi);
    }
    const unsigned short* wf = m ? wf1 : wf0;
#pragma unroll 1
    for (int nt = 0; nt < 8; ++nt) {
      v8f acc = {};
#pragma unroll
      for (int ks = 0; ks < 4; ++ks) {
        const uint4* p = (const uint4*)(wf + ((size_t)(ks * 8 + nt) * 32 + lane) * 16);
        acc = __builtin_amdgcn_wmma_f32_16x16x32_bf16(
            false, afr[ks], false, make_frag(p[0], p[1]), (short)0, acc, false, false);
      }
#pragma unroll
      for (int r = 0; r < 8; ++r) {
        int row = tile + r + lh * 8;
        int col = (m == 0) ? (nt * 16 + l16)
                           : (C_CH + (nt * 16 + l16) * 3 + (m - 1));
        size_t idx = (size_t)row * (4 * C_CH) + col;
        out[idx] = acc[r] + sc[idx];
      }
    }
  }
}

// ---------------------------------------------------------------------------
extern "C" void kernel_launch(void* const* d_in, const int* in_sizes, int n_in,
                              void* d_out, int out_size, void* d_ws, size_t ws_size,
                              hipStream_t stream) {
  (void)in_sizes; (void)n_in; (void)out_size; (void)ws_size;
  const float* node_feats = (const float*)d_in[0];
  const float* sc   = (const float*)d_in[1];
  const float* u3_s = (const float*)d_in[2];
  const float* u2_s = (const float*)d_in[3];
  const float* u1_s = (const float*)d_in[4];
  const float* u3_v = (const float*)d_in[5];
  const float* u2_v = (const float*)d_in[6];
  const float* u1_v = (const float*)d_in[7];
  const float* w3_s = (const float*)d_in[8];
  const float* w2_s = (const float*)d_in[9];
  const float* w1_s = (const float*)d_in[10];
  const float* w3_v = (const float*)d_in[11];
  const float* w2_v = (const float*)d_in[12];
  const float* w1_v = (const float*)d_in[13];
  const float* wl0  = (const float*)d_in[14];
  const float* wl1  = (const float*)d_in[15];
  float* out = (float*)d_out;

  char* ws = (char*)d_ws;
  unsigned short* gfrag = (unsigned short*)(ws);                 // 1,179,648 B
  unsigned short* wf0   = (unsigned short*)(ws + 1179648);       //    32,768 B
  unsigned short* wf1   = (unsigned short*)(ws + 1212416);       //    32,768 B
  float*          sv    = (float*)         (ws + 1245184);       // 16,777,216 B

  mace_pack_g_kernel<<<128, 256, 0, stream>>>(u3_s, u2_s, u1_s, u3_v, u2_v, u1_v,
                                              w3_s, w2_s, w1_s, w3_v, w2_v, w1_v, gfrag);
  mace_pack_wl_kernel<<<128, 256, 0, stream>>>(wl0, wl1, wf0, wf1);
  mace_contract_kernel<<<dim3(64, 128), 256, 0, stream>>>(node_feats, gfrag, sv);
  mace_linear_kernel<<<64, 256, 0, stream>>>(sv, wf0, wf1, sc, out);
}
